// GATEncoder_63788854280501
// MI455X (gfx1250) — compile-verified, matrix operand order checked by reading, not run on hardware
//
#include <hip/hip_runtime.h>
#include <hip/hip_bf16.h>

// ---------------- problem constants (fixed by the reference) ----------------
#define N_NODES 50000
#define E_EDGES 800000
#define IN_CH   128
#define HIDC    64
#define HEADS   4
#define HH      256               // HEADS*HIDC
#define ETOT    (E_EDGES + N_NODES)
#define OUTD    128               // 2*Z

typedef __bf16 bf16_t;
typedef bf16_t v16bf __attribute__((ext_vector_type(16)));
typedef bf16_t v8bf  __attribute__((ext_vector_type(8)));
typedef float  v8f   __attribute__((ext_vector_type(8)));

// ---------------- WMMA fragment loaders (CDNA5 16-bit 16x32 A layout) -------
// lanes 0-15: row = lane, K chunks [k0, k0+8) and [k0+16, k0+24)
// lanes 16-31: row = lane-16, K chunks [k0+8, k0+16) and [k0+24, k0+32)
__device__ __forceinline__ v16bf load_frag_16x32(const bf16_t* __restrict__ base,
                                                 int ld, int row0, int k0,
                                                 int lane, int rowMax) {
  int r = row0 + (lane & 15);
  if (r >= rowMax) r = rowMax - 1;                 // clamp (stores are guarded)
  int kb = k0 + ((lane >> 4) << 3);
  const v8bf* p0 = (const v8bf*)(base + (size_t)r * ld + kb);
  const v8bf* p1 = (const v8bf*)(base + (size_t)r * ld + kb + 16);
  v8bf lo = *p0, hi = *p1;
  v16bf v;
#pragma unroll
  for (int i = 0; i < 8; ++i) { v[i] = lo[i]; v[i + 8] = hi[i]; }
  return v;
}

// C = A[M][K](bf16) * Bt[256][K](bf16, pre-transposed weight) -> C[M][256] f32
// block = 256 threads = 8 waves arranged 2 (row) x 4 (col); tile 32x256.
__global__ __launch_bounds__(256) void gemm_bf16_wmma(
    const bf16_t* __restrict__ A, const bf16_t* __restrict__ Bt,
    float* __restrict__ C, int M, int K) {
  const int lane = threadIdx.x & 31;
  const int wave = threadIdx.x >> 5;
  const int row0 = blockIdx.x * 32 + (wave >> 2) * 16;
  const int col0 = (wave & 3) * 64;
  v8f acc[4] = {};
  for (int k0 = 0; k0 < K; k0 += 32) {
    v16bf a = load_frag_16x32(A, K, row0, k0, lane, M);
#pragma unroll
    for (int j = 0; j < 4; ++j) {
      v16bf b = load_frag_16x32(Bt, K, col0 + j * 16, k0, lane, HH);
      acc[j] = __builtin_amdgcn_wmma_f32_16x16x32_bf16(
          false, a, false, b, (short)0, acc[j], false, false);
    }
  }
  // C/D layout: lanes 0-15 -> col=lane, rows 0..7 ; lanes 16-31 -> rows 8..15
  const int ccol  = lane & 15;
  const int rbase = (lane >> 4) << 3;
#pragma unroll
  for (int j = 0; j < 4; ++j)
#pragma unroll
    for (int i = 0; i < 8; ++i) {
      int r = row0 + rbase + i;
      if (r < M) C[(size_t)r * HH + col0 + j * 16 + ccol] = acc[j][i];
    }
}

// ---------------- helpers ---------------------------------------------------
__global__ void fill_f32(float* __restrict__ p, float v, size_t n) {
  size_t t = (size_t)blockIdx.x * blockDim.x + threadIdx.x;
  if (t < n) p[t] = v;
}

__global__ void to_bf16(const float* __restrict__ x, bf16_t* __restrict__ y, size_t n) {
  size_t t = (size_t)blockIdx.x * blockDim.x + threadIdx.x;
  if (t < n) y[t] = (bf16_t)x[t];
}

// W[K][256] f32 -> Wt[256][Kpad] bf16 (zero-padded K rows)
__global__ void wtrans_bf16(const float* __restrict__ W, bf16_t* __restrict__ Wt,
                            int K, int Kpad) {
  int t = blockIdx.x * blockDim.x + threadIdx.x;
  if (t >= 256 * Kpad) return;
  int n = t / Kpad, k = t - n * Kpad;
  Wt[t] = (bf16_t)((k < K) ? W[(size_t)k * 256 + n] : 0.f);
}

// layer-2 input: [N][96] bf16 = [h(64) | l(1) | zeros(31)]
__global__ void build_a2(const float* __restrict__ h, const float* __restrict__ l,
                         bf16_t* __restrict__ a2) {
  int t = blockIdx.x * blockDim.x + threadIdx.x;
  if (t >= N_NODES * 96) return;
  int n = t / 96, k = t - n * 96;
  float v = (k < 64) ? h[(size_t)n * 64 + k] : (k == 64 ? l[n] : 0.f);
  a2[t] = (bf16_t)v;
}

__device__ __forceinline__ void atomicMaxF(float* addr, float val) {
  if (val >= 0.f) atomicMax((int*)addr, __float_as_int(val));
  else            atomicMin((unsigned int*)addr, __float_as_uint(val));
}

// ---------------- edge stage -------------------------------------------------
// pass A: e[edge][h] = sum_c leaky(xl[src]+xr[dst]) * att ; atomic max per dst
__global__ __launch_bounds__(256) void edge_attn_max(
    const float* __restrict__ xl, const float* __restrict__ xr,
    const long long* __restrict__ ei, const float* __restrict__ att,
    float* __restrict__ ebuf, float* __restrict__ mbuf) {
  int wv = blockIdx.x * 8 + (threadIdx.x >> 5);
  int lane = threadIdx.x & 31;
  if (wv >= ETOT) return;
  int src, dst;
  if (wv < E_EDGES) { src = (int)ei[wv]; dst = (int)ei[E_EDGES + wv]; }
  else              { src = wv - E_EDGES; dst = src; }
  const float* pl = xl + (size_t)src * HH;
  const float* pr = xr + (size_t)dst * HH;
  float eh[HEADS];
#pragma unroll
  for (int h = 0; h < HEADS; ++h) {
    float s = 0.f;
#pragma unroll
    for (int i = 0; i < 2; ++i) {
      int flat = h * 64 + i * 32 + lane;
      float v = pl[flat] + pr[flat];
      v = v > 0.f ? v : 0.2f * v;            // leaky relu, slope 0.2
      s += v * att[flat];
    }
#pragma unroll
    for (int off = 16; off > 0; off >>= 1) s += __shfl_xor(s, off, 32);
    eh[h] = s;
  }
  if (lane == 0) {
#pragma unroll
    for (int h = 0; h < HEADS; ++h) {
      ebuf[(size_t)wv * HEADS + h] = eh[h];
      atomicMaxF(&mbuf[(size_t)dst * HEADS + h], eh[h]);
    }
  }
}

// pass B: ebuf = exp(e - m[dst]); dbuf[dst] += ebuf
__global__ void edge_expsum(float* __restrict__ ebuf, const float* __restrict__ mbuf,
                            float* __restrict__ dbuf, const long long* __restrict__ ei) {
  size_t t = (size_t)blockIdx.x * blockDim.x + threadIdx.x;
  if (t >= (size_t)ETOT * HEADS) return;
  int e = (int)(t >> 2), h = (int)(t & 3);
  int dst = (e < E_EDGES) ? (int)ei[E_EDGES + e] : (e - E_EDGES);
  float a = __expf(ebuf[t] - mbuf[(size_t)dst * HEADS + h]);
  ebuf[t] = a;
  atomicAdd(&dbuf[(size_t)dst * HEADS + h], a);
}

// pass C: acc[dst] += xl[src] * alpha   (alpha = e / (d + 1e-16))
__global__ __launch_bounds__(256) void edge_message(
    const float* __restrict__ xl, const float* __restrict__ ebuf,
    const float* __restrict__ dbuf, const long long* __restrict__ ei,
    float* __restrict__ acc) {
  int wv = blockIdx.x * 8 + (threadIdx.x >> 5);
  int lane = threadIdx.x & 31;
  if (wv >= ETOT) return;
  int src, dst;
  if (wv < E_EDGES) { src = (int)ei[wv]; dst = (int)ei[E_EDGES + wv]; }
  else              { src = wv - E_EDGES; dst = src; }
  float alpha[HEADS];
#pragma unroll
  for (int h = 0; h < HEADS; ++h)
    alpha[h] = ebuf[(size_t)wv * HEADS + h] /
               (dbuf[(size_t)dst * HEADS + h] + 1e-16f);
  const float* pl = xl + (size_t)src * HH;
  float* pa = acc + (size_t)dst * HH;
#pragma unroll
  for (int i = 0; i < 8; ++i) {
    int flat = i * 32 + lane;
    atomicAdd(&pa[flat], pl[flat] * alpha[flat >> 6]);
  }
}

// mean over heads + bias -> h[N][64]
__global__ void mean_heads(const float* __restrict__ acc, const float* __restrict__ bias,
                           float* __restrict__ h) {
  size_t t = (size_t)blockIdx.x * blockDim.x + threadIdx.x;
  if (t >= (size_t)N_NODES * 64) return;
  int c = (int)(t & 63);
  const float* p = acc + ((t >> 6) * (size_t)HH) + c;
  h[t] = 0.25f * (p[0] + p[64] + p[128] + p[192]) + bias[c];
}

// ---------------- batchnorm + reductions ------------------------------------
__global__ __launch_bounds__(256) void col_stats(const float* __restrict__ X,
                                                 float* __restrict__ sums,
                                                 float* __restrict__ sumsq) {
  int t = blockIdx.x * blockDim.x + threadIdx.x;
  int c = t & 63;
  int stride = (gridDim.x * blockDim.x) >> 6;
  float s = 0.f, s2 = 0.f;
  for (int r = t >> 6; r < N_NODES; r += stride) {
    float v = X[(size_t)r * 64 + c];
    s += v; s2 += v * v;
  }
  atomicAdd(&sums[c], s);
  atomicAdd(&sumsq[c], s2);
}

__global__ void bn_relu(float* __restrict__ X, const float* __restrict__ sums,
                        const float* __restrict__ sumsq, const float* __restrict__ g,
                        const float* __restrict__ be) {
  size_t t = (size_t)blockIdx.x * blockDim.x + threadIdx.x;
  if (t >= (size_t)N_NODES * 64) return;
  int c = (int)(t & 63);
  float inv = 1.0f / (float)N_NODES;
  float mu  = sums[c] * inv;
  float var = sumsq[c] * inv - mu * mu;
  float v = (X[t] - mu) * rsqrtf(var + 1e-5f) * g[c] + be[c];
  X[t] = v > 0.f ? v : 0.f;
}

__global__ __launch_bounds__(256) void col_sum(const float* __restrict__ X,
                                               float* __restrict__ hbar) {
  int t = blockIdx.x * blockDim.x + threadIdx.x;
  int c = t & 63;
  int stride = (gridDim.x * blockDim.x) >> 6;
  float s = 0.f;
  for (int r = t >> 6; r < N_NODES; r += stride) s += X[(size_t)r * 64 + c];
  atomicAdd(&hbar[c], s);
}

// out[j] = blin[j] + sum_c (hbar[c]/N) * Wlin[c][j]   (mean commutes w/ linear)
__global__ void final_linear(const float* __restrict__ hbar,
                             const float* __restrict__ Wlin,
                             const float* __restrict__ blin,
                             float* __restrict__ out) {
  int j = threadIdx.x;                    // 128 threads
  float inv = 1.0f / (float)N_NODES;
  float s = blin[j];
#pragma unroll 8
  for (int c = 0; c < 64; ++c) s += (hbar[c] * inv) * Wlin[c * OUTD + j];
  out[j] = s;
}

// ---------------- host launch -----------------------------------------------
extern "C" void kernel_launch(void* const* d_in, const int* in_sizes, int n_in,
                              void* d_out, int out_size, void* d_ws, size_t ws_size,
                              hipStream_t stream) {
  const float*     x    = (const float*)d_in[0];
  const long long* ei   = (const long long*)d_in[1];
  const float*     l    = (const float*)d_in[2];
  const float*     Wl1  = (const float*)d_in[3];
  const float*     Wr1  = (const float*)d_in[4];
  const float*     att1 = (const float*)d_in[5];
  const float*     b1   = (const float*)d_in[6];
  const float*     g1   = (const float*)d_in[7];
  const float*     be1  = (const float*)d_in[8];
  const float*     Wl2  = (const float*)d_in[9];
  const float*     Wr2  = (const float*)d_in[10];
  const float*     att2 = (const float*)d_in[11];
  const float*     b2   = (const float*)d_in[12];
  const float*     g2   = (const float*)d_in[13];
  const float*     be2  = (const float*)d_in[14];
  const float*     Wlin = (const float*)d_in[15];
  const float*     blin = (const float*)d_in[16];
  float* out = (float*)d_out;

  char* base = (char*)d_ws;
  size_t off = 0;
  auto take = [&](size_t bytes) -> char* {
    char* p = base + off;
    off += (bytes + 255) & ~(size_t)255;
    return p;
  };
  float*  xl    = (float*) take((size_t)N_NODES * HH * 4);   // lin_l output
  float*  xracc = (float*) take((size_t)N_NODES * HH * 4);   // lin_r out, reused as msg acc
  float*  ebuf  = (float*) take((size_t)ETOT * HEADS * 4);
  float*  mbuf  = (float*) take((size_t)N_NODES * HEADS * 4);
  float*  dbuf  = (float*) take((size_t)N_NODES * HEADS * 4);
  float*  hbuf  = (float*) take((size_t)N_NODES * 64 * 4);   // layer act (reused l2)
  bf16_t* xb    = (bf16_t*)take((size_t)N_NODES * 128 * 2);  // bf16 input (reused l2)
  bf16_t* wta   = (bf16_t*)take((size_t)256 * 128 * 2);
  bf16_t* wtb   = (bf16_t*)take((size_t)256 * 128 * 2);
  float*  sums  = (float*) take(192 * 4);                    // sums|sumsq|hbar
  float*  sumsq = sums + 64;
  float*  hbar  = sums + 128;

  const int T = 256;
  auto blk  = [](size_t n, int t) { return (unsigned)((n + t - 1) / t); };
  const unsigned gEdgeWv = blk((size_t)ETOT, 8);             // wave-per-edge kernels
  const unsigned gN64    = blk((size_t)N_NODES * 64, T);
  const unsigned gN256   = blk((size_t)N_NODES * HH, T);
  const unsigned gNH     = blk((size_t)N_NODES * HEADS, T);
  const unsigned gEH     = blk((size_t)ETOT * HEADS, T);
  const unsigned gGemm   = blk((size_t)N_NODES, 32);

  // ---- layer 1 ----
  fill_f32<<<1, 192, 0, stream>>>(sums, 0.f, 192);
  to_bf16<<<blk((size_t)N_NODES * 128, T), T, 0, stream>>>(x, xb, (size_t)N_NODES * 128);
  wtrans_bf16<<<blk(256 * 128, T), T, 0, stream>>>(Wl1, wta, 128, 128);
  wtrans_bf16<<<blk(256 * 128, T), T, 0, stream>>>(Wr1, wtb, 128, 128);
  gemm_bf16_wmma<<<gGemm, T, 0, stream>>>(xb, wta, xl,    N_NODES, 128);
  gemm_bf16_wmma<<<gGemm, T, 0, stream>>>(xb, wtb, xracc, N_NODES, 128);
  fill_f32<<<gNH, T, 0, stream>>>(mbuf, -3.402823466e38f, (size_t)N_NODES * HEADS);
  fill_f32<<<gNH, T, 0, stream>>>(dbuf, 0.f, (size_t)N_NODES * HEADS);
  edge_attn_max<<<gEdgeWv, T, 0, stream>>>(xl, xracc, ei, att1, ebuf, mbuf);
  fill_f32<<<gN256, T, 0, stream>>>(xracc, 0.f, (size_t)N_NODES * HH);
  edge_expsum<<<gEH, T, 0, stream>>>(ebuf, mbuf, dbuf, ei);
  edge_message<<<gEdgeWv, T, 0, stream>>>(xl, ebuf, dbuf, ei, xracc);
  mean_heads<<<gN64, T, 0, stream>>>(xracc, b1, hbuf);
  col_stats<<<256, T, 0, stream>>>(hbuf, sums, sumsq);
  bn_relu<<<gN64, T, 0, stream>>>(hbuf, sums, sumsq, g1, be1);

  // ---- layer 2 (K = 65 padded to 96) ----
  build_a2<<<blk((size_t)N_NODES * 96, T), T, 0, stream>>>(hbuf, l, xb);
  wtrans_bf16<<<blk(256 * 96, T), T, 0, stream>>>(Wl2, wta, 65, 96);
  wtrans_bf16<<<blk(256 * 96, T), T, 0, stream>>>(Wr2, wtb, 65, 96);
  gemm_bf16_wmma<<<gGemm, T, 0, stream>>>(xb, wta, xl,    N_NODES, 96);
  gemm_bf16_wmma<<<gGemm, T, 0, stream>>>(xb, wtb, xracc, N_NODES, 96);
  fill_f32<<<gNH, T, 0, stream>>>(mbuf, -3.402823466e38f, (size_t)N_NODES * HEADS);
  fill_f32<<<gNH, T, 0, stream>>>(dbuf, 0.f, (size_t)N_NODES * HEADS);
  edge_attn_max<<<gEdgeWv, T, 0, stream>>>(xl, xracc, ei, att2, ebuf, mbuf);
  fill_f32<<<gN256, T, 0, stream>>>(xracc, 0.f, (size_t)N_NODES * HH);
  edge_expsum<<<gEH, T, 0, stream>>>(ebuf, mbuf, dbuf, ei);
  edge_message<<<gEdgeWv, T, 0, stream>>>(xl, ebuf, dbuf, ei, xracc);
  mean_heads<<<gN64, T, 0, stream>>>(xracc, b2, hbuf);
  fill_f32<<<1, 192, 0, stream>>>(sums, 0.f, 192);
  col_stats<<<256, T, 0, stream>>>(hbuf, sums, sumsq);
  bn_relu<<<gN64, T, 0, stream>>>(hbuf, sums, sumsq, g2, be2);

  // ---- final: mean over nodes, then tiny linear ----
  col_sum<<<256, T, 0, stream>>>(hbuf, hbar);
  final_linear<<<1, OUTD, 0, stream>>>(hbar, Wlin, blin, out);
}
